// WindowAttention_8177617731873
// MI455X (gfx1250) — matrix operand order repaired and checked by reading, not compiled, not run on hardware
//
#include <hip/hip_runtime.h>

// Swin window attention, MI455X (gfx1250), wave32 + V_WMMA_F32_16X16X32_BF16.
// Compute-bound (~4.4e11 FLOPs vs ~1.2GB HBM): convert once to bf16, do all
// four matmuls (QKV, QK^T, PV, proj) on the matrix engine with f32 accum.
// Round 3: explicit double-buffering of A/B fragments in the two K-loops so
// next-step loads are in flight while current WMMAs issue (kills the
// s_wait_loadcnt 0x0 full-drain before every WMMA seen in round 2).

#define DIMC  512
#define HEADS 16
#define NTOK  49
#define BWIN  4096
#define HD    32
#define NPAD  64

typedef __attribute__((ext_vector_type(16))) __bf16 v16bf;
typedef __attribute__((ext_vector_type(8)))  float  v8f;
typedef __attribute__((ext_vector_type(4)))  float  f32x4;

union Frag { f32x4 q[2]; v16bf bf; };
union Acc  { v8f v; float f[8]; };

__device__ __forceinline__ unsigned short f2bf(float x) {
  union { float f; unsigned u; } u; u.f = x;
  unsigned r = u.u + 0x7FFFu + ((u.u >> 16) & 1u);   // round-to-nearest-even
  return (unsigned short)(r >> 16);
}

__device__ __forceinline__ v8f wmma_bf16(const Frag& a, const Frag& b, v8f c) {
  return __builtin_amdgcn_wmma_f32_16x16x32_bf16(
      /*neg_a=*/false, a.bf, /*neg_b=*/false, b.bf,
      /*c_mod=*/(short)0, c, /*reuse_a=*/false, /*reuse_b=*/false);
}

// A fragment 16(m)x32(k) bf16 from row-major [m][k] source, stride ld elems.
// ISA layout: lane l -> m=l&15, h=l>>4; halves 0..7: K=8h+0..7, 8..15: K=16+8h+0..7
__device__ __forceinline__ Frag frag_a(const unsigned short* base, int ld) {
  int l = threadIdx.x & 31, m = l & 15, h = l >> 4;
  const unsigned short* p = base + (size_t)m * ld + 8 * h;
  Frag f;
  f.q[0] = *reinterpret_cast<const f32x4*>(p);
  f.q[1] = *reinterpret_cast<const f32x4*>(p + 16);
  return f;
}

// B fragment: logical B is 32(k)x16(n); source stored row-major [n][k] (B = src^T).
// ISA layout: lane l -> n=l&15, h=l>>4; halves 0..7: K=8h+0..7, 8..15: K=16+8h+0..7
__device__ __forceinline__ Frag frag_b(const unsigned short* base, int ld) {
  int l = threadIdx.x & 31, n = l & 15, h = l >> 4;
  const unsigned short* p = base + (size_t)n * ld + 8 * h;
  Frag f;
  f.q[0] = *reinterpret_cast<const f32x4*>(p);
  f.q[1] = *reinterpret_cast<const f32x4*>(p + 16);
  return f;
}

__global__ void cvt_kernel(const float* __restrict__ s, unsigned short* __restrict__ d,
                           size_t nn) {
  size_t i  = (size_t)blockIdx.x * blockDim.x + threadIdx.x;
  size_t st = (size_t)gridDim.x * blockDim.x;
  for (; i < nn; i += st) d[i] = f2bf(s[i]);
}

__global__ void bias_gather_kernel(const float* __restrict__ bt, const int* __restrict__ ri,
                                   float* __restrict__ bg) {
  int t = blockIdx.x * blockDim.x + threadIdx.x;
  if (t >= HEADS * NTOK * NTOK) return;
  int hh = t / (NTOK * NTOK), rem = t % (NTOK * NTOK);
  bg[t] = bt[ri[rem] * HEADS + hh];
}

// One block = one (window, head). 4 wave32s; wave = M-tile (row-block) in all phases.
__global__ __launch_bounds__(128) void attn_kernel(
    const unsigned short* __restrict__ xb,   // [BWIN][49][512] bf16
    const unsigned short* __restrict__ qw,   // [1536][512] bf16 (row-major [n][k])
    const float* __restrict__ qb,            // [1536]
    const float* __restrict__ biasg,         // [16][49][49] f32
    unsigned short* __restrict__ ao)         // [BWIN*49][512] bf16
{
  __shared__ alignas(16) unsigned short QS[NPAD * 40];  // q (scaled), [token][d], pad->40
  __shared__ alignas(16) unsigned short KS[NPAD * 40];  // k, [token][d]
  __shared__ alignas(16) unsigned short VT[HD * 72];    // v^T, [d][token], pad->72
  __shared__ alignas(16) unsigned short PS[NPAD * 72];  // softmax probs, [token][token]

  const int b = blockIdx.x, head = blockIdx.y;
  const int wave = (int)(threadIdx.x >> 5), lane = (int)(threadIdx.x & 31);
  const int n = lane & 15, h = lane >> 4;
  const int mt = wave;                       // row-block owned by this wave
  const unsigned short* xw = xb + (size_t)b * (NTOK * DIMC);
  const float scale = 0.17677669529663687f;  // 32^-0.5

  // ---- Phase 1: q,k,v = x @ W^T (+bias).
  // One A fragment (x row-block mt) feeds 6 WMMAs per k-step; A/B double-buffered.
  {
    const int wbase = head * HD;
    v8f z = {};
    Acc acc[6];
#pragma unroll
    for (int i = 0; i < 6; ++i) acc[i].v = z;
    int row = mt * 16 + (lane & 15); row = row > 48 ? 48 : row;  // pad 49->64 by clamp
    const unsigned short* pa = xw + (size_t)row * DIMC + 8 * h;
    const unsigned short* wb[6];
#pragma unroll
    for (int mat = 0; mat < 3; ++mat)
#pragma unroll
      for (int nt = 0; nt < 2; ++nt)
        wb[mat * 2 + nt] = qw + (size_t)(mat * DIMC + wbase + nt * 16) * DIMC;

    Frag a_cur, b_cur[6];
    a_cur.q[0] = *reinterpret_cast<const f32x4*>(pa);
    a_cur.q[1] = *reinterpret_cast<const f32x4*>(pa + 16);
#pragma unroll
    for (int i = 0; i < 6; ++i) b_cur[i] = frag_b(wb[i], DIMC);

#pragma unroll
    for (int k0 = 0; k0 < DIMC; k0 += 32) {
      // issue next k-step's loads before consuming current fragments
      int kn = (k0 + 32 < DIMC) ? k0 + 32 : 0;  // last-iter loads are dead
      Frag a_nxt, b_nxt[6];
      a_nxt.q[0] = *reinterpret_cast<const f32x4*>(pa + kn);
      a_nxt.q[1] = *reinterpret_cast<const f32x4*>(pa + kn + 16);
#pragma unroll
      for (int i = 0; i < 6; ++i) b_nxt[i] = frag_b(wb[i] + kn, DIMC);
#pragma unroll
      for (int i = 0; i < 6; ++i) acc[i].v = wmma_bf16(a_cur, b_cur[i], acc[i].v);
      a_cur = a_nxt;
#pragma unroll
      for (int i = 0; i < 6; ++i) b_cur[i] = b_nxt[i];
    }
#pragma unroll
    for (int nt = 0; nt < 2; ++nt) {
      float bq = qb[0 * DIMC + wbase + nt * 16 + n];
      float bk = qb[1 * DIMC + wbase + nt * 16 + n];
      float bv = qb[2 * DIMC + wbase + nt * 16 + n];
#pragma unroll
      for (int r = 0; r < 8; ++r) {
        int mrow = mt * 16 + r + 8 * h;
        QS[mrow * 40 + nt * 16 + n] = f2bf((acc[0 + nt].f[r] + bq) * scale);
        KS[mrow * 40 + nt * 16 + n] = f2bf(acc[2 + nt].f[r] + bk);
        VT[(nt * 16 + n) * 72 + mrow] = f2bf(acc[4 + nt].f[r] + bv);  // v stored transposed
      }
    }
  }
  __syncthreads();

  // ---- Phase 2: S = q k^T (+bias), softmax -> PS. K=32 == head_dim (one WMMA/tile).
  {
    Frag aq = frag_a(QS + mt * 16 * 40, 40);
    Acc s[4];
#pragma unroll
    for (int c = 0; c < 4; ++c) {
      v8f z = {};
      Frag bk = frag_b(KS + c * 16 * 40, 40);
      s[c].v = wmma_bf16(aq, bk, z);
    }
    const float* bh = biasg + head * (NTOK * NTOK);
    float p[4][8];
#pragma unroll
    for (int c = 0; c < 4; ++c)
#pragma unroll
      for (int r = 0; r < 8; ++r) {
        int i = mt * 16 + r + 8 * h, j = c * 16 + n;
        float v = s[c].f[r];
        if (j < NTOK) { if (i < NTOK) v += bh[i * NTOK + j]; }
        else v = -1e30f;               // mask padded key columns
        p[c][r] = v;
      }
    // row softmax: row m = r + 8h lives entirely in one 16-lane half at VGPR r
#pragma unroll
    for (int r = 0; r < 8; ++r) {
      float rm = fmaxf(fmaxf(p[0][r], p[1][r]), fmaxf(p[2][r], p[3][r]));
#pragma unroll
      for (int off = 1; off < 16; off <<= 1) rm = fmaxf(rm, __shfl_xor(rm, off, 32));
      float e0 = __expf(p[0][r] - rm), e1 = __expf(p[1][r] - rm);
      float e2 = __expf(p[2][r] - rm), e3 = __expf(p[3][r] - rm);
      float sum = e0 + e1 + e2 + e3;
#pragma unroll
      for (int off = 1; off < 16; off <<= 1) sum += __shfl_xor(sum, off, 32);
      float inv = 1.0f / sum;
      p[0][r] = e0 * inv; p[1][r] = e1 * inv; p[2][r] = e2 * inv; p[3][r] = e3 * inv;
    }
#pragma unroll
    for (int c = 0; c < 4; ++c)
#pragma unroll
      for (int r = 0; r < 8; ++r)
        PS[(mt * 16 + r + 8 * h) * 72 + c * 16 + n] = f2bf(p[c][r]);
  }
  __syncthreads();

  // ---- Phase 3: O = P @ V (via V^T). K = 64 padded tokens (pad probs are exact 0).
  {
    v8f z = {};
    Acc o0, o1; o0.v = z; o1.v = z;
#pragma unroll
    for (int ks = 0; ks < 2; ++ks) {
      Frag ap = frag_a(PS + (size_t)mt * 16 * 72 + ks * 32, 72);
      Frag b0 = frag_b(VT + ks * 32, 72);
      Frag b1 = frag_b(VT + 16 * 72 + ks * 32, 72);
      o0.v = wmma_bf16(ap, b0, o0.v);
      o1.v = wmma_bf16(ap, b1, o1.v);
    }
#pragma unroll
    for (int r = 0; r < 8; ++r) {
      int token = mt * 16 + r + 8 * h;
      if (token < NTOK) {
        size_t rowoff = ((size_t)b * NTOK + token) * DIMC + head * HD;
        ao[rowoff + n]      = f2bf(o0.f[r]);
        ao[rowoff + 16 + n] = f2bf(o1.f[r]);
      }
    }
  }
}

// Projection GEMM: out[200704][512] = AO(bf16) @ proj_w^T + proj_b, f32 out.
// 64x128 tile per block, 8 waves, 4 output tiles per wave; A/B double-buffered.
__global__ __launch_bounds__(256) void proj_kernel(
    const unsigned short* __restrict__ AO,   // [200704][512] bf16
    const unsigned short* __restrict__ PW,   // [512][512] bf16 (row-major [n][k])
    const float* __restrict__ pb,            // [512]
    float* __restrict__ out)                 // [200704][512] f32
{
  const int m0 = blockIdx.x * 64, n0 = blockIdx.y * 128;
  const int wave = (int)(threadIdx.x >> 5), lane = (int)(threadIdx.x & 31);
  const int mt = wave & 3, ntb = (wave >> 2) * 4;
  v8f z = {};
  Acc o[4];
#pragma unroll
  for (int t = 0; t < 4; ++t) o[t].v = z;
  const unsigned short* arow = AO + (size_t)(m0 + mt * 16) * DIMC;
  const unsigned short* wrow[4];
#pragma unroll
  for (int t = 0; t < 4; ++t) wrow[t] = PW + (size_t)(n0 + (ntb + t) * 16) * DIMC;

  Frag a_cur = frag_a(arow, DIMC);
  Frag b_cur[4];
#pragma unroll
  for (int t = 0; t < 4; ++t) b_cur[t] = frag_b(wrow[t], DIMC);

#pragma unroll
  for (int k0 = 0; k0 < DIMC; k0 += 32) {
    int kn = (k0 + 32 < DIMC) ? k0 + 32 : 0;  // last-iter loads are dead
    Frag a_nxt = frag_a(arow + kn, DIMC);
    Frag b_nxt[4];
#pragma unroll
    for (int t = 0; t < 4; ++t) b_nxt[t] = frag_b(wrow[t] + kn, DIMC);
#pragma unroll
    for (int t = 0; t < 4; ++t) o[t].v = wmma_bf16(a_cur, b_cur[t], o[t].v);
    a_cur = a_nxt;
#pragma unroll
    for (int t = 0; t < 4; ++t) b_cur[t] = b_nxt[t];
  }

  const int n = lane & 15, h = lane >> 4;
#pragma unroll
  for (int t = 0; t < 4; ++t)
#pragma unroll
    for (int r = 0; r < 8; ++r) {
      int row = m0 + mt * 16 + r + 8 * h;
      int c = n0 + (ntb + t) * 16 + n;
      out[(size_t)row * DIMC + c] = o[t].f[r] + pb[c];
    }
}

extern "C" void kernel_launch(void* const* d_in, const int* in_sizes, int n_in,
                              void* d_out, int out_size, void* d_ws, size_t ws_size,
                              hipStream_t stream) {
  (void)in_sizes; (void)n_in; (void)out_size; (void)ws_size;
  const float* x    = (const float*)d_in[0];
  const float* qkvw = (const float*)d_in[1];
  const float* qkvb = (const float*)d_in[2];
  const float* pw   = (const float*)d_in[3];
  const float* pb   = (const float*)d_in[4];
  const float* bt   = (const float*)d_in[5];
  const int*   ri   = (const int*)d_in[6];
  float* out = (float*)d_out;

  char* ws = (char*)d_ws;
  size_t off = 0;
  unsigned short* xb  = (unsigned short*)(ws + off); off += (size_t)BWIN * NTOK * DIMC * 2;
  unsigned short* qwb = (unsigned short*)(ws + off); off += (size_t)3 * DIMC * DIMC * 2;
  unsigned short* pwb = (unsigned short*)(ws + off); off += (size_t)DIMC * DIMC * 2;
  unsigned short* ao  = (unsigned short*)(ws + off); off += (size_t)BWIN * NTOK * DIMC * 2;
  float* biasg = (float*)(ws + off);                 off += (size_t)HEADS * NTOK * NTOK * 4;

  cvt_kernel<<<dim3(4096), dim3(256), 0, stream>>>(x, xb, (size_t)BWIN * NTOK * DIMC);
  cvt_kernel<<<dim3(256),  dim3(256), 0, stream>>>(qkvw, qwb, (size_t)3 * DIMC * DIMC);
  cvt_kernel<<<dim3(64),   dim3(256), 0, stream>>>(pw, pwb, (size_t)DIMC * DIMC);
  bias_gather_kernel<<<dim3((HEADS * NTOK * NTOK + 255) / 256), dim3(256), 0, stream>>>(bt, ri, biasg);
  attn_kernel<<<dim3(BWIN, HEADS), dim3(128), 0, stream>>>(xb, qwb, qkvb, biasg, ao);
  proj_kernel<<<dim3((BWIN * NTOK) / 64, DIMC / 128), dim3(256), 0, stream>>>(ao, pwb, pb, out);
}
